// HypergraphEncoder_1838246002961
// MI455X (gfx1250) — compile-verified, matrix operand order checked by reading, not run on hardware
//
#include <hip/hip_runtime.h>
#include <hip/hip_bf16.h>

#define N_NODES 100000
#define N_EDGES 20000
#define NNZ_CNT 1600000

typedef __attribute__((ext_vector_type(16))) __bf16 v16bf;
typedef __attribute__((ext_vector_type(8)))  float  v8f;

union FragBF { unsigned short u[16]; v16bf v; };

__device__ __forceinline__ unsigned short f2bf(float f) {
    unsigned int u = __float_as_uint(f);
    // round-to-nearest-even bf16 truncation
    unsigned int r = (u + 0x7FFFu + ((u >> 16) & 1u)) >> 16;
    return (unsigned short)r;
}

// ---------------------------------------------------------------------------
// Degree computation + inversion
// ---------------------------------------------------------------------------
__global__ void k_degrees(const int* __restrict__ node_idx,
                          const int* __restrict__ edge_idx,
                          float* __restrict__ Bdeg, float* __restrict__ Ddeg,
                          int nnz) {
    int i = blockIdx.x * blockDim.x + threadIdx.x;
    if (i >= nnz) return;
    atomicAdd(&Bdeg[edge_idx[i]], 1.0f);
    atomicAdd(&Ddeg[node_idx[i]], 1.0f);
}

__global__ void k_invert(float* __restrict__ buf, int n) {
    int i = blockIdx.x * blockDim.x + threadIdx.x;
    if (i >= n) return;
    float v = buf[i];
    buf[i] = (v > 0.0f) ? (1.0f / v) : 0.0f;
}

// ---------------------------------------------------------------------------
// WMMA bf16 GEMM: C[M,N] = A[M,K] * W[K,N]  (fp32 in, bf16 WMMA, fp32 out)
// Block = 256 threads = 8 waves. Tile: BM=128 (16 rows/wave), BN=64, BK=32.
// A tile staged raw fp32 via GLOBAL_LOAD_ASYNC_TO_LDS_B128 (ASYNCcnt),
// converted to bf16 at fragment build. W tile staged sync + transposed.
// ---------------------------------------------------------------------------
#define BM 128
#define BN 64
#define BK 32

__global__ __launch_bounds__(256)
void k_gemm_bf16(const float* __restrict__ A, const float* __restrict__ W,
                 float* __restrict__ C, int M, int N, int K) {
    __shared__ float          sA[BM][BK + 4];   // raw fp32 [m][k], +4 pad (bank spread)
    __shared__ unsigned short lW[BN][BK + 4];   // bf16 transposed [n][k]

    const int tid  = threadIdx.x;
    const int lane = tid & 31;
    const int wv   = tid >> 5;
    const int m0   = blockIdx.x * BM;
    const int n0   = blockIdx.y * BN;

    v8f acc0 = {}, acc1 = {}, acc2 = {}, acc3 = {};

    const int row = wv * 16 + (lane & 15);   // A-frag row within block tile
    const int kb  = (lane >> 4) * 8;         // K sub-offset per ISA 16-bit A layout

    for (int k0 = 0; k0 < K; k0 += BK) {
        // --- Async-stage A tile (128 rows x 32 fp32): one B128 per lane issue ---
        #pragma unroll
        for (int i = 0; i < 4; ++i) {
            int idx = i * 256 + tid;
            int r   = idx >> 3;              // 0..127
            int c   = (idx & 7) << 2;        // float col: 0,4,...,28
            int gm  = m0 + r;  gm = (gm < M) ? gm : (M - 1);       // row clamp (dup ok)
            int gk  = k0 + c;  gk = (gk + 3 < K) ? gk : (K - 4);   // chunk clamp (masked later)
            unsigned long long ga = (unsigned long long)(A + (size_t)gm * K + gk);
            unsigned lo = (unsigned)(size_t)&sA[r][c];
            asm volatile("global_load_async_to_lds_b128 %0, %1, off"
                         :: "v"(lo), "v"(ga) : "memory");
        }
        // --- Sync-stage W tile (32x64) -> bf16, transposed [n][k]; branch-free ---
        #pragma unroll
        for (int idx = tid; idx < BK * BN; idx += 256) {
            int kk  = idx >> 6;
            int n   = idx & 63;
            int gk  = k0 + kk;
            int gks = (gk < K) ? gk : (K - 1);
            float v = W[(size_t)gks * N + (n0 + n)];
            lW[n][kk] = (gk < K) ? f2bf(v) : (unsigned short)0;
        }
        asm volatile("s_wait_asynccnt 0x0" ::: "memory");
        __syncthreads();

        // Build A fragment (16x32): elems 0-7 -> K=kb..kb+7, 8-15 -> K=16+kb..
        FragBF af;
        #pragma unroll
        for (int e = 0; e < 8; ++e) {
            int k1 = kb + e, k2 = 16 + kb + e;
            float a1 = sA[row][k1];
            float a2 = sA[row][k2];
            af.u[e]     = (k0 + k1 < K) ? f2bf(a1) : (unsigned short)0;
            af.u[e + 8] = (k0 + k2 < K) ? f2bf(a2) : (unsigned short)0;
        }

        v8f* accs[4] = { &acc0, &acc1, &acc2, &acc3 };
        #pragma unroll
        for (int nt = 0; nt < 4; ++nt) {
            FragBF bfm;
            int col = nt * 16 + (lane & 15);
            #pragma unroll
            for (int e = 0; e < 8; ++e) {
                bfm.u[e]     = lW[col][kb + e];
                bfm.u[e + 8] = lW[col][16 + kb + e];
            }
            *accs[nt] = __builtin_amdgcn_wmma_f32_16x16x32_bf16(
                false, af.v, false, bfm.v, (short)0, *accs[nt], false, false);
        }
        __syncthreads();
    }

    // Store D: VGPR v holds row v (lanes 0-15) / row v+8 (lanes 16-31), col = lane%16
    const int colb = lane & 15;
    const int rsel = (lane >> 4) * 8;
    v8f accs[4] = { acc0, acc1, acc2, acc3 };
    #pragma unroll
    for (int nt = 0; nt < 4; ++nt) {
        int gc = n0 + nt * 16 + colb;
        #pragma unroll
        for (int v = 0; v < 8; ++v) {
            int gr = m0 + wv * 16 + rsel + v;
            if (gr < M && gc < N) C[(size_t)gr * N + gc] = accs[nt][v];
        }
    }
}

// ---------------------------------------------------------------------------
// Scatter-add: dst[sidx[i]] += src[gidx[i]]  (rows of F floats, float4 chunks)
// Targets are L2-resident (e: 20.5 MB, xw: 102 MB vs 192 MB L2).
// ---------------------------------------------------------------------------
__global__ void k_scatter(const float* __restrict__ src, float* __restrict__ dst,
                          const int* __restrict__ gidx, const int* __restrict__ sidx,
                          int nnz, int F) {
    const int chunks = F >> 2;
    long long t = (long long)blockIdx.x * blockDim.x + threadIdx.x;
    long long total = (long long)nnz * chunks;
    if (t >= total) return;
    int i = (int)(t / chunks);
    int c = (int)(t % chunks) << 2;
    int g = gidx[i], s = sidx[i];
    const float4 v = *(const float4*)(src + (size_t)g * F + c);
    float* d = dst + (size_t)s * F + c;
    atomicAdd(d + 0, v.x);
    atomicAdd(d + 1, v.y);
    atomicAdd(d + 2, v.z);
    atomicAdd(d + 3, v.w);
}

__global__ void k_scale_rows(float* __restrict__ buf, const float* __restrict__ inv,
                             int rows, int F) {
    long long t = (long long)blockIdx.x * blockDim.x + threadIdx.x;
    if (t >= (long long)rows * F) return;
    int r = (int)(t / F);
    buf[t] *= inv[r];
}

__global__ void k_finish(float* __restrict__ buf, const float* __restrict__ inv,
                         const float* __restrict__ bias, int rows, int F, int relu) {
    long long t = (long long)blockIdx.x * blockDim.x + threadIdx.x;
    if (t >= (long long)rows * F) return;
    int r = (int)(t / F);
    int f = (int)(t % F);
    float v = buf[t] * inv[r] + bias[f];
    if (relu) v = (v > 0.0f) ? v : 0.01f * v;
    buf[t] = v;
}

// ---------------------------------------------------------------------------
// Attention pooling: logits -> softmax over N_NODES -> weighted sum (128)
// ---------------------------------------------------------------------------
__global__ void k_logits(const float* __restrict__ h, const float* __restrict__ aw,
                         const float* __restrict__ ab, float* __restrict__ logits,
                         float* __restrict__ blockmax, int n) {
    __shared__ float smax[256];
    int i = blockIdx.x * 256 + threadIdx.x;
    float l = -1e30f;
    if (i < n) {
        float s = ab[0];
        #pragma unroll 4
        for (int f = 0; f < 128; ++f) s += h[(size_t)i * 128 + f] * aw[f];
        logits[i] = s;
        l = s;
    }
    smax[threadIdx.x] = l;
    __syncthreads();
    for (int o = 128; o > 0; o >>= 1) {
        if (threadIdx.x < o)
            smax[threadIdx.x] = fmaxf(smax[threadIdx.x], smax[threadIdx.x + o]);
        __syncthreads();
    }
    if (threadIdx.x == 0) blockmax[blockIdx.x] = smax[0];
}

__global__ void k_reduce_max(const float* __restrict__ bm, int nb,
                             float* __restrict__ gmax) {
    __shared__ float s[256];
    float m = -1e30f;
    for (int i = threadIdx.x; i < nb; i += 256) m = fmaxf(m, bm[i]);
    s[threadIdx.x] = m;
    __syncthreads();
    for (int o = 128; o > 0; o >>= 1) {
        if (threadIdx.x < o) s[threadIdx.x] = fmaxf(s[threadIdx.x], s[threadIdx.x + o]);
        __syncthreads();
    }
    if (threadIdx.x == 0) gmax[0] = s[0];
}

__global__ void k_attn_acc(const float* __restrict__ h, const float* __restrict__ logits,
                           const float* __restrict__ gmax, float* __restrict__ acc,
                           float* __restrict__ gsum, int n) {
    __shared__ float lacc[2][128];
    __shared__ float lsum[256];
    const int f = threadIdx.x & 127;
    const int sub = threadIdx.x >> 7;
    lacc[sub][f] = 0.0f;
    float wsum = 0.0f;
    const int start = blockIdx.x * 256;
    const int end = min(start + 256, n);
    const float m = gmax[0];
    __syncthreads();
    for (int nd = start + sub; nd < end; nd += 2) {
        float w = __expf(logits[nd] - m);
        if (f == 0) wsum += w;
        lacc[sub][f] += w * h[(size_t)nd * 128 + f];
    }
    lsum[threadIdx.x] = wsum;
    __syncthreads();
    if (sub == 0) {
        lacc[0][f] += lacc[1][f];
        atomicAdd(&acc[f], lacc[0][f]);
    }
    for (int o = 128; o > 0; o >>= 1) {
        if (threadIdx.x < o) lsum[threadIdx.x] += lsum[threadIdx.x + o];
        __syncthreads();
    }
    if (threadIdx.x == 0) atomicAdd(gsum, lsum[0]);
}

__global__ void k_out(const float* __restrict__ acc, const float* __restrict__ gsum,
                      float* __restrict__ out) {
    int f = threadIdx.x;
    if (f < 128) out[f] = acc[f] / gsum[0];
}

// ---------------------------------------------------------------------------
// Host orchestration
// ---------------------------------------------------------------------------
extern "C" void kernel_launch(void* const* d_in, const int* in_sizes, int n_in,
                              void* d_out, int out_size, void* d_ws, size_t ws_size,
                              hipStream_t stream) {
    const float* x       = (const float*)d_in[0];
    const int*   hedge   = (const int*)d_in[1];
    const float* W1      = (const float*)d_in[2];
    const float* b1      = (const float*)d_in[3];
    const float* W2      = (const float*)d_in[4];
    const float* b2      = (const float*)d_in[5];
    const float* W3      = (const float*)d_in[6];
    const float* b3      = (const float*)d_in[7];
    const float* attn_W  = (const float*)d_in[8];
    const float* attn_b  = (const float*)d_in[9];

    const int nnz = in_sizes[1] / 2;
    const int* node_idx = hedge;
    const int* edge_idx = hedge + nnz;

    // Workspace carve-out (256B aligned)
    size_t off = 0;
    auto carve = [&](size_t bytes) -> void* {
        off = (off + 255) & ~(size_t)255;
        void* p = (char*)d_ws + off;
        off += bytes;
        return p;
    };
    float* bufA     = (float*)carve((size_t)N_NODES * 256 * 4);  // 102.4 MB
    float* bufB     = (float*)carve((size_t)N_NODES * 256 * 4);  // 102.4 MB
    float* bufE     = (float*)carve((size_t)N_EDGES * 256 * 4);  // 20.5 MB
    float* Binv     = (float*)carve((size_t)N_EDGES * 4);
    float* Dinv     = (float*)carve((size_t)N_NODES * 4);
    float* logits   = (float*)carve((size_t)N_NODES * 4);
    float* blockmax = (float*)carve(4096);
    float* gmax     = (float*)carve(4);
    float* gsum     = (float*)carve(4);
    float* acc      = (float*)carve(128 * 4);

    // --- degrees ---
    hipMemsetAsync(Binv, 0, (size_t)N_EDGES * 4, stream);
    hipMemsetAsync(Dinv, 0, (size_t)N_NODES * 4, stream);
    k_degrees<<<(nnz + 255) / 256, 256, 0, stream>>>(node_idx, edge_idx, Binv, Dinv, nnz);
    k_invert<<<(N_EDGES + 255) / 256, 256, 0, stream>>>(Binv, N_EDGES);
    k_invert<<<(N_NODES + 255) / 256, 256, 0, stream>>>(Dinv, N_NODES);

    auto run_hconv = [&](const float* in, int K, const float* Wm, const float* bias,
                         float* out, int F, int relu) {
        dim3 g((N_NODES + BM - 1) / BM, F / BN);
        k_gemm_bf16<<<g, 256, 0, stream>>>(in, Wm, out, N_NODES, F, K);
        // node -> edge
        hipMemsetAsync(bufE, 0, (size_t)N_EDGES * F * 4, stream);
        long long tot = (long long)nnz * (F / 4);
        int sblk = (int)((tot + 255) / 256);
        k_scatter<<<sblk, 256, 0, stream>>>(out, bufE, node_idx, edge_idx, nnz, F);
        k_scale_rows<<<((long long)N_EDGES * F + 255) / 256, 256, 0, stream>>>(bufE, Binv, N_EDGES, F);
        // edge -> node (out buffer reused: xw fully consumed above)
        hipMemsetAsync(out, 0, (size_t)N_NODES * F * 4, stream);
        k_scatter<<<sblk, 256, 0, stream>>>(bufE, out, edge_idx, node_idx, nnz, F);
        k_finish<<<((long long)N_NODES * F + 255) / 256, 256, 0, stream>>>(out, Dinv, bias, N_NODES, F, relu);
    };

    run_hconv(x,    300, W1, b1, bufA, 256, 1);   // layer 1 + leaky relu
    run_hconv(bufA, 256, W2, b2, bufB, 256, 1);   // layer 2 + leaky relu
    run_hconv(bufB, 256, W3, b3, bufA, 128, 0);   // layer 3 (no relu)

    // --- attention pooling over nodes ---
    int nb = (N_NODES + 255) / 256;
    k_logits<<<nb, 256, 0, stream>>>(bufA, attn_W, attn_b, logits, blockmax, N_NODES);
    k_reduce_max<<<1, 256, 0, stream>>>(blockmax, nb, gmax);
    hipMemsetAsync(acc, 0, 128 * 4, stream);
    hipMemsetAsync(gsum, 0, 4, stream);
    k_attn_acc<<<nb, 256, 0, stream>>>(bufA, logits, gmax, acc, gsum, N_NODES);
    k_out<<<1, 128, 0, stream>>>(acc, gsum, (float*)d_out);
}
